// PerceiverLayer_59760174956727
// MI455X (gfx1250) — compile-verified, matrix operand order checked by reading, not run on hardware
//
#include <hip/hip_runtime.h>
#include <hip/hip_bf16.h>

typedef __bf16 bf16;
typedef __attribute__((ext_vector_type(16))) __bf16 v16bf;
typedef __attribute__((ext_vector_type(8)))  __bf16 v8bf;
typedef __attribute__((ext_vector_type(8)))  float  v8f;
typedef __attribute__((__vector_size__(4 * sizeof(int)))) int v4i;

// ---- optional CDNA5 async global->LDS staging (ASYNCcnt path) ----
#if defined(__has_builtin)
#  if __has_builtin(__builtin_amdgcn_global_load_async_to_lds_b128) && \
      __has_builtin(__builtin_amdgcn_s_wait_asynccnt)
#    define USE_ASYNC_LDS 1
#  endif
#endif
#ifndef USE_ASYNC_LDS
#  define USE_ASYNC_LDS 0
#endif

// ---------------- problem constants ----------------
constexpr int  Bc  = 4;
constexpr int  Tc  = 2048;
constexpr int  Dc  = 1024;
constexpr int  HDc = 64;
constexpr int  Ic  = 4096;
constexpr int  Kc  = 4;
constexpr float EPSc = 1e-6f;

constexpr long long E_ACT = (long long)Bc * Tc * Dc;   // 8,388,608
constexpr long long E_ATT = (long long)Bc * Tc * Tc;   // 16,777,216
constexpr long long E_DD  = (long long)Dc * Dc;        // 1,048,576
constexpr long long E_FI  = (long long)Dc * 2 * Ic;    // 8,388,608
constexpr long long E_FC  = (long long)Ic * Dc;        // 4,194,304
constexpr long long E_GLU = (long long)Bc * Tc * Ic;   // 33,554,432

// ---------------- GEMM tiling ----------------
#define BM 128
#define BN 64
#define BK 64
#define LDT (BK + 8)   // 72 elems -> 144B pitch (multiple of 16B)

constexpr unsigned SM_AS_OFF  = 0;
constexpr unsigned SM_BS_OFF  = (unsigned)(BM * LDT * 2);              // 18432
constexpr unsigned SM_BS2_OFF = SM_BS_OFF + (unsigned)(BN * LDT * 2);  // 27648
constexpr unsigned SM_BYTES_NOGLU = SM_BS2_OFF;                        // 27648
constexpr unsigned SM_BYTES_GLU   = SM_BS2_OFF + (unsigned)(BN * LDT * 2);  // 36864

#if USE_ASYNC_LDS
// copy 16 bytes global->LDS via the async engine; ldsoff is a byte offset into
// the (dynamic) LDS allocation, which starts at 0 for these kernels.
__device__ inline void async_cp16(const bf16* g, unsigned ldsoff) {
  __builtin_amdgcn_global_load_async_to_lds_b128(
      (__attribute__((address_space(1))) v4i*)(uintptr_t)g,
      (__attribute__((address_space(3))) v4i*)(uintptr_t)ldsoff, 0, 0);
}
#endif

// Per-lane bf16 fragment (A 16x32 or Bt 16x32): two contiguous 16B runs.
//   lane L: row = L%16, half = L/16
//   elems  0..7  = tile[row][ks*32 + half*8      .. +8)
//   elems  8..15 = tile[row][ks*32 + 16 + half*8 .. +8)
// -> the documented 16-bit A/B VGPR packing, loaded as 2x ds_load_b128.
__device__ inline v16bf frag_ld(const bf16* __restrict__ p, int half) {
  const v8bf lo = *(const v8bf*)(p + half * 8);
  const v8bf hi = *(const v8bf*)(p + 16 + half * 8);
  return __builtin_shufflevector(lo, hi, 0, 1, 2, 3, 4, 5, 6, 7, 8, 9, 10, 11, 12, 13, 14, 15);
}

// D = A(bf16) x B^T(bf16) + bias, fp32 accumulate via v_wmma_f32_16x16x32_bf16.
// B is ALWAYS row-major [N x Kd] (weights pre-transposed; K naturally so;
// V written transposed by the CBT epilogue of its producing GEMM).
// Block tile 128x64, 8 waves, each wave owns a 32x32 tile (2x2 WMMA).
// GLU: second B panel (r-half); epilogue emits bf16( (z+bz) * silu(r+br) ).
// CBT: write Cb transposed ([N x M], leading dim ldc) -- used to produce V^T.
template <bool GLU, bool CBT>
__global__ __launch_bounds__(256) void gemm_wmma_kernel(
    const bf16* __restrict__ Ag, const bf16* __restrict__ Bg,
    const bf16* __restrict__ B2g, const float* __restrict__ bias,
    const float* __restrict__ bias2, float* __restrict__ Cf,
    bf16* __restrict__ Cb, int Kd, int lda, int ldb, int ldc,
    long long sA, long long sB, long long sC) {
  extern __shared__ char dynsmem[];
  bf16* As  = (bf16*)(dynsmem + SM_AS_OFF);
  bf16* Bs  = (bf16*)(dynsmem + SM_BS_OFF);
  bf16* Bs2 = (bf16*)(dynsmem + SM_BS2_OFF);

  const int tid  = threadIdx.x;
  const int lane = tid & 31;
  const int wave = tid >> 5;
  const int mi = wave & 3;   // 32-row band within the 128-row tile
  const int ni = wave >> 2;  // 32-col band within the 64-col tile

  const long long z = blockIdx.z;
  const long long blockMr = (long long)blockIdx.y * BM;
  const long long blockNr = (long long)blockIdx.x * BN;

  Ag += z * sA;
  Bg += z * sB;
  if constexpr (GLU) B2g += z * sB;
  if (Cf) Cf += z * sC;
  if (Cb) Cb += z * sC;

  v8f acc00 = {}, acc01 = {}, acc10 = {}, acc11 = {};
  v8f racc00 = {}, racc01 = {}, racc10 = {}, racc11 = {};

  // staging maps (256 threads):
  // A tile 128x64: thread -> row = tid>>1, 32 contiguous bf16 at (tid&1)*32
  const int ar = tid >> 1;
  const int ac = (tid & 1) * 32;
  // B tile 64x64: thread -> row = tid>>2, 16 contiguous bf16 at (tid&3)*16
  const int br = tid >> 2;
  const int bc = (tid & 3) * 16;

  const int mrow = lane & 15;
  const int half = lane >> 4;

  for (int kt = 0; kt < Kd; kt += BK) {
    const bf16* asrc = Ag + (blockMr + ar) * (long long)lda + kt + ac;
    const bf16* bsrc = Bg + (blockNr + br) * (long long)ldb + kt + bc;
#if USE_ASYNC_LDS
    {
#pragma unroll
      for (int j = 0; j < 4; ++j)
        async_cp16(asrc + j * 8, SM_AS_OFF + (unsigned)(ar * LDT + ac + j * 8) * 2);
      async_cp16(bsrc,     SM_BS_OFF + (unsigned)(br * LDT + bc) * 2);
      async_cp16(bsrc + 8, SM_BS_OFF + (unsigned)(br * LDT + bc + 8) * 2);
      if constexpr (GLU) {
        const bf16* b2src = B2g + (blockNr + br) * (long long)ldb + kt + bc;
        async_cp16(b2src,     SM_BS2_OFF + (unsigned)(br * LDT + bc) * 2);
        async_cp16(b2src + 8, SM_BS2_OFF + (unsigned)(br * LDT + bc + 8) * 2);
      }
      __builtin_amdgcn_s_wait_asynccnt(0);
    }
#else
    {
#pragma unroll
      for (int j = 0; j < 4; ++j)
        *(uint4*)&As[ar * LDT + ac + j * 8] = *(const uint4*)(asrc + j * 8);
      *(uint4*)&Bs[br * LDT + bc]     = *(const uint4*)bsrc;
      *(uint4*)&Bs[br * LDT + bc + 8] = *(const uint4*)(bsrc + 8);
      if constexpr (GLU) {
        const bf16* b2src = B2g + (blockNr + br) * (long long)ldb + kt + bc;
        *(uint4*)&Bs2[br * LDT + bc]     = *(const uint4*)b2src;
        *(uint4*)&Bs2[br * LDT + bc + 8] = *(const uint4*)(b2src + 8);
      }
      if (kt + BK < Kd) {  // global_prefetch_b8 of next K tile
        __builtin_prefetch(asrc + BK, 0, 1);
        __builtin_prefetch(bsrc + BK, 0, 1);
      }
    }
#endif
    __syncthreads();

#pragma unroll
    for (int ks = 0; ks < BK / 32; ++ks) {
      const v16bf a0 = frag_ld(As + (mi * 32 + mrow) * LDT + ks * 32, half);
      const v16bf a1 = frag_ld(As + (mi * 32 + 16 + mrow) * LDT + ks * 32, half);
      const v16bf b0 = frag_ld(Bs + (ni * 32 + mrow) * LDT + ks * 32, half);
      const v16bf b1 = frag_ld(Bs + (ni * 32 + 16 + mrow) * LDT + ks * 32, half);
      acc00 = __builtin_amdgcn_wmma_f32_16x16x32_bf16(false, a0, false, b0, (short)0, acc00, false, false);
      acc01 = __builtin_amdgcn_wmma_f32_16x16x32_bf16(false, a0, false, b1, (short)0, acc01, false, false);
      acc10 = __builtin_amdgcn_wmma_f32_16x16x32_bf16(false, a1, false, b0, (short)0, acc10, false, false);
      acc11 = __builtin_amdgcn_wmma_f32_16x16x32_bf16(false, a1, false, b1, (short)0, acc11, false, false);
      if constexpr (GLU) {
        const v16bf c0 = frag_ld(Bs2 + (ni * 32 + mrow) * LDT + ks * 32, half);
        const v16bf c1 = frag_ld(Bs2 + (ni * 32 + 16 + mrow) * LDT + ks * 32, half);
        racc00 = __builtin_amdgcn_wmma_f32_16x16x32_bf16(false, a0, false, c0, (short)0, racc00, false, false);
        racc01 = __builtin_amdgcn_wmma_f32_16x16x32_bf16(false, a0, false, c1, (short)0, racc01, false, false);
        racc10 = __builtin_amdgcn_wmma_f32_16x16x32_bf16(false, a1, false, c0, (short)0, racc10, false, false);
        racc11 = __builtin_amdgcn_wmma_f32_16x16x32_bf16(false, a1, false, c1, (short)0, racc11, false, false);
      }
    }
    __syncthreads();
  }

  // C/D layout: lane n = L%16, rows r + 8*(L/16)
  const int n_in = lane & 15;
  const int mh   = lane >> 4;
#pragma unroll
  for (int ms = 0; ms < 2; ++ms) {
    const v8f& az = ms ? acc10 : acc00;
    const v8f& bz = ms ? acc11 : acc01;
    const v8f& ar_ = ms ? racc10 : racc00;
    const v8f& br_ = ms ? racc11 : racc01;
#pragma unroll
    for (int r = 0; r < 8; ++r) {
      const long long m  = blockMr + mi * 32 + ms * 16 + r + 8 * mh;
      const long long n0 = blockNr + ni * 32 + n_in;
      const long long n1 = n0 + 16;
      if constexpr (!GLU) {
        const float v0 = az[r] + (bias ? bias[n0] : 0.0f);
        const float v1 = bz[r] + (bias ? bias[n1] : 0.0f);
        if (Cf) { Cf[m * ldc + n0] = v0; Cf[m * ldc + n1] = v1; }
        if (Cb) {
          if constexpr (CBT) {
            Cb[n0 * ldc + m] = (bf16)v0;
            Cb[n1 * ldc + m] = (bf16)v1;
          } else {
            Cb[m * ldc + n0] = (bf16)v0;
            Cb[m * ldc + n1] = (bf16)v1;
          }
        }
      } else {
        const float z0 = az[r] + bias[n0], z1 = bz[r] + bias[n1];
        const float r0 = ar_[r] + bias2[n0], r1 = br_[r] + bias2[n1];
        const float s0 = r0 / (1.0f + __expf(-r0));
        const float s1 = r1 / (1.0f + __expf(-r1));
        Cb[m * ldc + n0] = (bf16)(z0 * s0);
        Cb[m * ldc + n1] = (bf16)(z1 * s1);
      }
    }
  }
}

// ---------------- helpers ----------------
__device__ inline float block_sum256(float v) {
  __shared__ float red[8];
  for (int o = 16; o > 0; o >>= 1) v += __shfl_down(v, o, 32);
  if ((threadIdx.x & 31) == 0) red[threadIdx.x >> 5] = v;
  __syncthreads();
  if (threadIdx.x == 0) {
    float t = 0.f;
    for (int i = 0; i < 8; ++i) t += red[i];
    red[0] = t;
  }
  __syncthreads();
  return red[0];
}

// fp32 [Kd x N] -> bf16 [N x Kd] (tiled transpose through LDS)
__global__ __launch_bounds__(256) void cast_transpose_kernel(
    const float* __restrict__ x, bf16* __restrict__ y, int Kd, int N) {
  __shared__ float tile[32][33];
  const int n0 = blockIdx.x * 32;
  const int k0 = blockIdx.y * 32;
  const int tx = threadIdx.x & 31;
  const int ty = threadIdx.x >> 5;  // 0..7
#pragma unroll
  for (int j = 0; j < 32; j += 8)
    tile[ty + j][tx] = x[(long long)(k0 + ty + j) * N + n0 + tx];
  __syncthreads();
#pragma unroll
  for (int j = 0; j < 32; j += 8)
    y[(long long)(n0 + ty + j) * Kd + k0 + tx] = (bf16)tile[tx][ty + j];
}

// rmsnorm over last dim (D=1024), block per row, bf16 output
__global__ __launch_bounds__(256) void rmsnorm_cast_kernel(
    const float* __restrict__ X, const float* __restrict__ g, bf16* __restrict__ Y) {
  const long long row = blockIdx.x;
  const float* x = X + row * Dc;
  float ss = 0.f;
  for (int d = threadIdx.x; d < Dc; d += 256) { const float v = x[d]; ss += v * v; }
  const float tot = block_sum256(ss);
  const float sc = rsqrtf(tot / (float)Dc + EPSc);
  for (int d = threadIdx.x; d < Dc; d += 256) Y[row * Dc + d] = (bf16)(x[d] * sc * g[d]);
}

// RoPE on first HD=64 channels only (matches reference), cast to bf16
__global__ void rope_cast_kernel(const float* __restrict__ X, bf16* __restrict__ Y) {
  long long i = (long long)blockIdx.x * blockDim.x + threadIdx.x;
  const long long stride = (long long)gridDim.x * blockDim.x;
  for (; i < E_ACT; i += stride) {
    const int d = (int)(i % Dc);
    const long long r = i / Dc;
    const int t = (int)(r % Tc);
    const float x = X[i];
    float y;
    if (d < HDc) {
      const int pair = d >> 1;
      const float inv = __powf(10000.0f, -((float)(2 * pair) / (float)HDc));
      const float f = (float)t * inv;
      float s, c;
      __sincosf(f, &s, &c);
      if (d & 1) y = x * c + X[i - 1] * s;
      else       y = x * c - X[i + 1] * s;
    } else {
      y = x;
    }
    Y[i] = (bf16)y;
  }
}

// causal mask, +h state, h_out, exp, row-sum, *2/sum -> bf16 attention weights
__global__ __launch_bounds__(256) void softmax_state_kernel(
    const float* __restrict__ S, const float* __restrict__ Hin,
    float* __restrict__ Hout, bf16* __restrict__ Aout) {
  const long long row = blockIdx.x;          // 0 .. B*T-1
  const int q = (int)(row % Tc);
  const float* s = S + row * (long long)Tc;
  const float* hi = Hin + row * (long long)Tc;
  float* ho = Hout + row * (long long)Tc;
  bf16* ao = Aout + row * (long long)Tc;
  float loc[Tc / 256];
  float sum = 0.f;
#pragma unroll
  for (int j = 0; j < Tc / 256; ++j) {
    const int col = threadIdx.x + j * 256;
    float st;
    if (col > q) st = -__builtin_inff();
    else st = s[col] * 0.125f + hi[col];     // 1/sqrt(HD)=0.125
    ho[col] = st;
    const float e = __expf(st);
    loc[j] = e;
    sum += e;
  }
  const float tot = block_sum256(sum);
  const float inv = 2.0f / tot;
#pragma unroll
  for (int j = 0; j < Tc / 256; ++j) {
    const int col = threadIdx.x + j * 256;
    ao[col] = (bf16)(loc[j] * inv);
  }
}

// depthwise causal conv, K=4
__global__ void conv_kernel(const float* __restrict__ latC, const float* __restrict__ cst,
                            const float* __restrict__ w, const float* __restrict__ b,
                            float* __restrict__ y) {
  long long i = (long long)blockIdx.x * blockDim.x + threadIdx.x;
  const long long stride = (long long)gridDim.x * blockDim.x;
  for (; i < E_ACT; i += stride) {
    const int d = (int)(i % Dc);
    const long long r = i / Dc;
    const int t = (int)(r % Tc);
    const int bb = (int)(r / Tc);
    float acc = b[d];
#pragma unroll
    for (int j = 0; j < Kc; ++j) {
      const int idx = t + 1 + j;  // index into concat(state, x^T)
      float v;
      if (idx < Kc) v = cst[((long long)bb * Dc + d) * Kc + idx];
      else          v = latC[((long long)bb * Tc + (idx - Kc)) * Dc + d];
      acc += w[d * Kc + j] * v;
    }
    y[i] = acc;
  }
}

__global__ void conv_state_out_kernel(const float* __restrict__ latC, float* __restrict__ cso) {
  const int i = blockIdx.x * blockDim.x + threadIdx.x;  // B*D*K
  if (i >= Bc * Dc * Kc) return;
  const int j = i % Kc;
  const int d = (i / Kc) % Dc;
  const int b = i / (Dc * Kc);
  cso[i] = latC[((long long)b * Tc + (Tc - Kc + j)) * Dc + d];
}

__global__ void add3_kernel(const float* __restrict__ a, const float* __restrict__ b,
                            const float* __restrict__ c, float* __restrict__ y, long long n) {
  long long i = (long long)blockIdx.x * blockDim.x + threadIdx.x;
  const long long stride = (long long)gridDim.x * blockDim.x;
  for (; i < n; i += stride) y[i] = a[i] + b[i] + c[i];
}

__global__ void add2_cast_kernel(const float* __restrict__ a, const float* __restrict__ b,
                                 float* __restrict__ yf, bf16* __restrict__ yb, long long n) {
  long long i = (long long)blockIdx.x * blockDim.x + threadIdx.x;
  const long long stride = (long long)gridDim.x * blockDim.x;
  for (; i < n; i += stride) {
    const float v = a[i] + b[i];
    yf[i] = v;
    yb[i] = (bf16)v;
  }
}

// ---------------- host-side launch machinery ----------------
static void launch_gemm(hipStream_t st, const bf16* A, const bf16* B, const bf16* B2,
                        const float* bias, const float* bias2, float* Cf, bf16* Cb,
                        int M, int N, int Kd, int lda, int ldb, int ldc, int batch,
                        long long sA, long long sB, long long sC, bool glu, bool cbt) {
  dim3 grid(N / BN, M / BM, batch), block(256);
  if (glu)
    gemm_wmma_kernel<true, false><<<grid, block, SM_BYTES_GLU, st>>>(A, B, B2, bias, bias2, Cf, Cb, Kd, lda, ldb, ldc, sA, sB, sC);
  else if (cbt)
    gemm_wmma_kernel<false, true><<<grid, block, SM_BYTES_NOGLU, st>>>(A, B, nullptr, bias, nullptr, Cf, Cb, Kd, lda, ldb, ldc, sA, sB, sC);
  else
    gemm_wmma_kernel<false, false><<<grid, block, SM_BYTES_NOGLU, st>>>(A, B, nullptr, bias, nullptr, Cf, Cb, Kd, lda, ldb, ldc, sA, sB, sC);
}

static inline int ew_blocks(long long n) {
  long long b = (n + 255) / 256;
  return (int)(b > 16384 ? 16384 : b);
}

struct AttnScratch {
  float *Qf, *Kf, *FCf;
  bf16 *Qb, *Kb, *Vt, *AOb, *Ab;
  float* S;
};

static void run_attention(hipStream_t st, const bf16* qin_b, const bf16* kv_b,
                          const bf16* wq, const float* bq, const bf16* wk, const float* bk,
                          const bf16* wv, const float* bv, const bf16* wfc, const float* bfc,
                          const float* h_in, float* h_out, const AttnScratch& s) {
  const long long MT = (long long)Bc * Tc;
  const long long sTD = (long long)Tc * Dc;
  // Q,K projections (flat)
  launch_gemm(st, qin_b, wq, nullptr, bq, nullptr, s.Qf, nullptr, (int)MT, Dc, Dc, Dc, Dc, Dc, 1, 0, 0, 0, false, false);
  launch_gemm(st, kv_b, wk, nullptr, bk, nullptr, s.Kf, nullptr, (int)MT, Dc, Dc, Dc, Dc, Dc, 1, 0, 0, 0, false, false);
  // V projection, batched, transposed epilogue -> Vt [B][D][T]
  launch_gemm(st, kv_b, wv, nullptr, bv, nullptr, nullptr, s.Vt, Tc, Dc, Dc, Dc, Dc, Tc,
              Bc, sTD, 0, sTD, false, true);
  // RoPE + cast
  rope_cast_kernel<<<ew_blocks(E_ACT), 256, 0, st>>>(s.Qf, s.Qb);
  rope_cast_kernel<<<ew_blocks(E_ACT), 256, 0, st>>>(s.Kf, s.Kb);
  // scores = Q K^T (batched; K already [T x D] row-major = B^T layout)
  launch_gemm(st, s.Qb, s.Kb, nullptr, nullptr, nullptr, s.S, nullptr, Tc, Tc, Dc, Dc, Dc, Tc,
              Bc, sTD, sTD, (long long)Tc * Tc, false, false);
  // state/softmax
  softmax_state_kernel<<<Bc * Tc, 256, 0, st>>>(s.S, h_in, h_out, s.Ab);
  // attn @ V  (batched; B operand is Vt [D x T] row-major = B^T layout)
  launch_gemm(st, s.Ab, s.Vt, nullptr, nullptr, nullptr, nullptr, s.AOb, Tc, Dc, Tc, Tc, Tc, Dc,
              Bc, (long long)Tc * Tc, sTD, sTD, false, false);
  // fc projection (flat)
  launch_gemm(st, s.AOb, wfc, nullptr, bfc, nullptr, s.FCf, nullptr, (int)MT, Dc, Dc, Dc, Dc, Dc, 1, 0, 0, 0, false, false);
}

extern "C" void kernel_launch(void* const* d_in, const int* in_sizes, int n_in,
                              void* d_out, int out_size, void* d_ws, size_t ws_size,
                              hipStream_t stream) {
  (void)in_sizes; (void)n_in; (void)out_size; (void)ws_size;

  // ---- inputs (setup_inputs() recursive insertion order) ----
  const float* x_lat_in   = (const float*)d_in[0];
  const float* x_inp_in   = (const float*)d_in[1];
  const float* x_outp_in  = (const float*)d_in[2];
  const float* h_in       = (const float*)d_in[3];   // (3,B,T,T)
  const float* cst_in     = (const float*)d_in[4];   // (B,D,K)
  const float* latent_rms = (const float*)d_in[5];
  const float* cross_rms  = (const float*)d_in[6];
  const float* glu_rms    = (const float*)d_in[7];
  const float* out_rms    = (const float*)d_in[8];
  const float* cproj_w = (const float*)d_in[9];  const float* cproj_b = (const float*)d_in[10];
  const float* iproj_w = (const float*)d_in[11]; const float* iproj_b = (const float*)d_in[12];
  const float* oproj_w = (const float*)d_in[13]; const float* oproj_b = (const float*)d_in[14];
  const float* ca_qw = (const float*)d_in[15]; const float* ca_qb = (const float*)d_in[16];
  const float* ca_kw = (const float*)d_in[17]; const float* ca_kb = (const float*)d_in[18];
  const float* ca_vw = (const float*)d_in[19]; const float* ca_vb = (const float*)d_in[20];
  const float* ca_fw = (const float*)d_in[21]; const float* ca_fb = (const float*)d_in[22];
  const float* oa_qw = (const float*)d_in[23]; const float* oa_qb = (const float*)d_in[24];
  const float* oa_kw = (const float*)d_in[25]; const float* oa_kb = (const float*)d_in[26];
  const float* oa_vw = (const float*)d_in[27]; const float* oa_vb = (const float*)d_in[28];
  const float* oa_fw = (const float*)d_in[29]; const float* oa_fb = (const float*)d_in[30];
  const float* conv_w = (const float*)d_in[31]; const float* conv_b = (const float*)d_in[32];
  const float* fi_w = (const float*)d_in[33]; const float* fi_b = (const float*)d_in[34];
  const float* fc_w = (const float*)d_in[35]; const float* fc_b = (const float*)d_in[36];

  // ---- outputs: (lat, out, h_stack, conv_state) concatenated ----
  float* out_lat = (float*)d_out;
  float* out_out = out_lat + E_ACT;
  float* out_h   = out_out + E_ACT;
  float* out_cst = out_h + 3 * E_ATT;

  // ---- workspace carve-up ----
  char* ws = (char*)d_ws;
  size_t off = 0;
  auto alloc = [&](long long bytes) -> void* {
    void* p = ws + off;
    off += (size_t)((bytes + 255) & ~255LL);
    return p;
  };
  // all weights stored TRANSPOSED bf16: [dout x din]
  bf16* w_cproj = (bf16*)alloc(E_DD * 2);
  bf16* w_iproj = (bf16*)alloc(E_DD * 2);
  bf16* w_oproj = (bf16*)alloc(E_DD * 2);
  bf16* w_caq = (bf16*)alloc(E_DD * 2); bf16* w_cak = (bf16*)alloc(E_DD * 2);
  bf16* w_cav = (bf16*)alloc(E_DD * 2); bf16* w_caf = (bf16*)alloc(E_DD * 2);
  bf16* w_oaq = (bf16*)alloc(E_DD * 2); bf16* w_oak = (bf16*)alloc(E_DD * 2);
  bf16* w_oav = (bf16*)alloc(E_DD * 2); bf16* w_oaf = (bf16*)alloc(E_DD * 2);
  bf16* w_fi  = (bf16*)alloc(E_FI * 2);   // [2I x D]
  bf16* w_fc  = (bf16*)alloc(E_FC * 2);   // [D x I]

  bf16*  nb   = (bf16*)alloc(E_ACT * 2);   // rmsnorm output (reused)
  float* xl_f = (float*)alloc(E_ACT * 4);  bf16* xl_b = (bf16*)alloc(E_ACT * 2);
  float* xi_f = (float*)alloc(E_ACT * 4);  bf16* xi_b = (bf16*)alloc(E_ACT * 2);
  float* xo_f = (float*)alloc(E_ACT * 4);  bf16* xo_b = (bf16*)alloc(E_ACT * 2);

  AttnScratch as;
  as.Qf  = (float*)alloc(E_ACT * 4);
  as.Kf  = (float*)alloc(E_ACT * 4);
  as.Qb  = (bf16*)alloc(E_ACT * 2);
  as.Kb  = (bf16*)alloc(E_ACT * 2);
  as.Vt  = (bf16*)alloc(E_ACT * 2);        // [B][D][T]
  as.S   = (float*)alloc(E_ATT * 4);
  as.Ab  = (bf16*)alloc(E_ATT * 2);
  as.AOb = (bf16*)alloc(E_ACT * 2);
  as.FCf = (float*)alloc(E_ACT * 4);

  float* latC  = (float*)alloc(E_ACT * 4);  // pre-conv lat
  float* latv  = (float*)alloc(E_ACT * 4);  // post-conv lat
  bf16*  Hb    = (bf16*)alloc(E_GLU * 2);   // z*silu(r)
  float* gfc   = (float*)alloc(E_ACT * 4);  // glu_fc output
  bf16*  latFb = (bf16*)alloc(E_ACT * 2);   // final lat, bf16 (attn2 kv input)

  // ---- convert+transpose all weights to bf16 [N x Kd] ----
  struct { const float* s; bf16* d; int Kd, N; } wjobs[] = {
      {cproj_w, w_cproj, Dc, Dc}, {iproj_w, w_iproj, Dc, Dc}, {oproj_w, w_oproj, Dc, Dc},
      {ca_qw, w_caq, Dc, Dc}, {ca_kw, w_cak, Dc, Dc}, {ca_vw, w_cav, Dc, Dc}, {ca_fw, w_caf, Dc, Dc},
      {oa_qw, w_oaq, Dc, Dc}, {oa_kw, w_oak, Dc, Dc}, {oa_vw, w_oav, Dc, Dc}, {oa_fw, w_oaf, Dc, Dc},
      {fi_w, w_fi, Dc, 2 * Ic}, {fc_w, w_fc, Ic, Dc}};
  for (auto& j : wjobs) {
    dim3 g(j.N / 32, j.Kd / 32);
    cast_transpose_kernel<<<g, 256, 0, stream>>>(j.s, j.d, j.Kd, j.N);
  }

  const long long MT = (long long)Bc * Tc;  // 8192 rows

  // ---- x_latent / x_input / x_out projections (fused rmsnorm -> bf16 -> GEMM) ----
  rmsnorm_cast_kernel<<<(int)MT, 256, 0, stream>>>(x_lat_in, latent_rms, nb);
  launch_gemm(stream, nb, w_cproj, nullptr, cproj_b, nullptr, xl_f, xl_b, (int)MT, Dc, Dc, Dc, Dc, Dc, 1, 0, 0, 0, false, false);
  rmsnorm_cast_kernel<<<(int)MT, 256, 0, stream>>>(x_inp_in, cross_rms, nb);
  launch_gemm(stream, nb, w_iproj, nullptr, iproj_b, nullptr, xi_f, xi_b, (int)MT, Dc, Dc, Dc, Dc, Dc, 1, 0, 0, 0, false, false);
  rmsnorm_cast_kernel<<<(int)MT, 256, 0, stream>>>(x_outp_in, out_rms, nb);
  launch_gemm(stream, nb, w_oproj, nullptr, oproj_b, nullptr, xo_f, xo_b, (int)MT, Dc, Dc, Dc, Dc, Dc, 1, 0, 0, 0, false, false);

  // ---- cross attention (h slab 0) ----
  run_attention(stream, xl_b, xi_b, w_caq, ca_qb, w_cak, ca_kb, w_cav, ca_vb, w_caf, ca_fb,
                h_in, out_h, as);
  add3_kernel<<<ew_blocks(E_ACT), 256, 0, stream>>>(as.FCf, xl_f, xi_f, latC, E_ACT);

  // ---- causal depthwise conv ----
  conv_kernel<<<ew_blocks(E_ACT), 256, 0, stream>>>(latC, cst_in, conv_w, conv_b, latv);
  conv_state_out_kernel<<<(Bc * Dc * Kc + 255) / 256, 256, 0, stream>>>(latC, out_cst);

  // ---- GLU (fused dual-panel GEMM with silu epilogue) ----
  rmsnorm_cast_kernel<<<(int)MT, 256, 0, stream>>>(latv, glu_rms, nb);
  launch_gemm(stream, nb, w_fi, w_fi + (long long)Ic * Dc, fi_b, fi_b + Ic, nullptr, Hb,
              (int)MT, Ic, Dc, Dc, Dc, Ic, 1, 0, 0, 0, true, false);
  launch_gemm(stream, Hb, w_fc, nullptr, fc_b, nullptr, gfc, nullptr,
              (int)MT, Dc, Ic, Ic, Ic, Dc, 1, 0, 0, 0, false, false);
  add2_cast_kernel<<<ew_blocks(E_ACT), 256, 0, stream>>>(gfc, latv, out_lat, latFb, E_ACT);

  // ---- out attention (q from x_out, kv from final lat; h slab 2) ----
  run_attention(stream, xo_b, latFb, w_oaq, oa_qb, w_oak, oa_kb, w_oav, oa_vb, w_oaf, oa_fb,
                h_in + 2 * E_ATT, out_h + 2 * E_ATT, as);
  add3_kernel<<<ew_blocks(E_ACT), 256, 0, stream>>>(as.FCf, xo_f, xi_f, out_out, E_ACT);

  // ---- h2 passes through unchanged ----
  (void)hipMemcpyAsync(out_h + E_ATT, h_in + E_ATT, (size_t)E_ATT * sizeof(float),
                       hipMemcpyDeviceToDevice, stream);
}
